// BLSTMGermEval_56607668961824
// MI455X (gfx1250) — matrix operand order repaired and checked by reading, not compiled, not run on hardware
//
#include <hip/hip_runtime.h>
#include <hip/hip_bf16.h>

// Problem constants
#define B_    64
#define S_    256
#define E_    300
#define H_    512
#define C_    25
#define NG    2048      // 4*H
#define EPAD  320       // E padded to multiple of 32
#define K0TOT 832       // EPAD + H
#define K1TOT 1536      // 2H + H

typedef __attribute__((ext_vector_type(16))) __bf16 v16bf;
typedef __attribute__((ext_vector_type(8)))  float  v8f;

union Frag16 { v16bf v; uint4 q[2]; };

static __device__ __forceinline__ v8f wmma_bf16(v16bf a, v16bf b, v8f c) {
  // D = A(16x32 bf16) * B(32x16 bf16) + C(16x16 f32)
  return __builtin_amdgcn_wmma_f32_16x16x32_bf16(false, a, false, b, (short)0, c,
                                                 false, false);
}

static __device__ __forceinline__ float sigmoidf_(float x) {
  return 1.0f / (1.0f + __expf(-x));
}
static __device__ __forceinline__ float tanhf_(float x) {
  x = fminf(15.f, fmaxf(-15.f, x));
  float e = __expf(2.0f * x);
  return (e - 1.0f) / (e + 1.0f);
}

// ---------------------------------------------------------------------------
// 1) Embedding gather + pad to bf16, time-major: xpad[s][b][EPAD]
// ---------------------------------------------------------------------------
__global__ void embed_pad_kernel(const int* __restrict__ ids,
                                 const float* __restrict__ emb,
                                 __bf16* __restrict__ xpad) {
  int bid = blockIdx.x;          // = s*64 + b
  int s = bid >> 6, b = bid & 63;
  int id = ids[b * S_ + s];
  const float* src = emb + (size_t)id * E_;
  __bf16* dst = xpad + (size_t)bid * EPAD;
  for (int j = threadIdx.x; j < EPAD; j += blockDim.x)
    dst[j] = (__bf16)(j < E_ ? src[j] : 0.0f);
}

// ---------------------------------------------------------------------------
// 2) Weight convert: w[f32, (Korig_x+H) x 2048] -> wt[bf16, 2048 x ktot],
//    transposed (column-major over N) with K padded (zero rows inserted).
// ---------------------------------------------------------------------------
__global__ void wconv_kernel(const float* __restrict__ w,
                             __bf16* __restrict__ wt,
                             int korigx, int kpadx, int ktot) {
  int n = blockIdx.x;            // 0..2047
  for (int k = threadIdx.x; k < ktot; k += blockDim.x) {
    int ko = (k < kpadx) ? ((k < korigx) ? k : -1) : (k - (kpadx - korigx));
    float v = (ko >= 0) ? w[(size_t)ko * NG + n] : 0.0f;
    wt[(size_t)n * ktot + k] = (__bf16)v;
  }
}

// ---------------------------------------------------------------------------
// Per-direction grid barrier (16 blocks), agent scope.
// bar[0] = arrival counter, bar[1] = generation.
// ---------------------------------------------------------------------------
static __device__ __forceinline__ void dir_barrier(unsigned* bar, int nblk) {
  __syncthreads();
  if (threadIdx.x == 0) {
    unsigned* cnt = bar;
    unsigned* gen = bar + 1;
    unsigned g = __hip_atomic_load(gen, __ATOMIC_RELAXED, __HIP_MEMORY_SCOPE_AGENT);
    unsigned a = __hip_atomic_fetch_add(cnt, 1u, __ATOMIC_ACQ_REL, __HIP_MEMORY_SCOPE_AGENT);
    if (a == (unsigned)(nblk - 1)) {
      __hip_atomic_store(cnt, 0u, __ATOMIC_RELAXED, __HIP_MEMORY_SCOPE_AGENT);
      __hip_atomic_fetch_add(gen, 1u, __ATOMIC_RELEASE, __HIP_MEMORY_SCOPE_AGENT);
    } else {
      while (__hip_atomic_load(gen, __ATOMIC_ACQUIRE, __HIP_MEMORY_SCOPE_AGENT) == g)
        __builtin_amdgcn_s_sleep(1);
    }
  }
  __syncthreads();
}

// ---------------------------------------------------------------------------
// 3) Persistent bidirectional LSTM layer.
//    Grid: 32 blocks (blocks 0-15 = forward, 16-31 = backward), 256 thr each.
//    Each wave owns one (Mtile[16 rows], 16 hidden units) cell for all steps:
//    c-state in registers, h-state via global bf16 [64][512] (L2 resident).
//    WMMA: A = [x_t || h_{t-1}] rows (per-lane row loads -> per-batch reversed
//    positions for the bw direction come for free), B = Wt columns.
//    K loop is split into the x-region and h-region so the hot loop is pure
//    pointer-increment loads + WMMA (no per-tile region select).
// ---------------------------------------------------------------------------
__global__ __launch_bounds__(256) void lstm_layer_kernel(
    const __bf16* __restrict__ Xin,       // [S][64][Kx]
    const __bf16* __restrict__ WtFw,      // [2048][Ktot]
    const __bf16* __restrict__ WtBw,
    const float* __restrict__ bFw, const float* __restrict__ bBw,
    const int* __restrict__ lens,
    __bf16* __restrict__ hFw, __bf16* __restrict__ hBw,   // [64][512]
    unsigned* barFw, unsigned* barBw,
    __bf16* __restrict__ Out,             // [S][64][2H], bf16
    int Kx, int Ktot) {
  const int dir = blockIdx.x >> 4;
  const __bf16* Wt  = dir ? WtBw : WtFw;
  const float* bias = dir ? bBw  : bFw;
  __bf16* hSt       = dir ? hBw  : hFw;
  unsigned* bar     = dir ? barBw : barFw;
  const int colOff  = dir * H_;

  const int lane = threadIdx.x & 31;
  const int wave = threadIdx.x >> 5;
  const int blk  = blockIdx.x & 15;
  const int p    = blk * 8 + wave;        // 0..127
  const int mtile = p & 3;                // 4 M tiles of 16 rows
  const int htile = p >> 2;               // 32 hidden tiles of 16 units
  const int hi = lane >> 4, ln = lane & 15;

  const int mA   = mtile * 16 + ln;       // A-fragment row for this lane
  const int lenA = lens[mA];
  const int n    = htile * 16 + ln;       // hidden unit (B-fragment column)

  const float bi  = bias[n];
  const float bj  = bias[H_ + n];
  const float bff = bias[2 * H_ + n] + 1.0f;  // forget bias folded in
  const float bo  = bias[3 * H_ + n];

  int   len_r[8];
  float c_r[8];
#pragma unroll
  for (int r = 0; r < 8; ++r) {
    len_r[r] = lens[mtile * 16 + hi * 8 + r];
    c_r[r] = 0.0f;
  }

  // Per-lane weight base pointers (hi selects the 16-row K half of the B tile)
  const __bf16* const wbase0 = Wt + ((size_t)(0 * H_ + n)) * Ktot + hi * 16;
  const __bf16* const wbase1 = Wt + ((size_t)(1 * H_ + n)) * Ktot + hi * 16;
  const __bf16* const wbase2 = Wt + ((size_t)(2 * H_ + n)) * Ktot + hi * 16;
  const __bf16* const wbase3 = Wt + ((size_t)(3 * H_ + n)) * Ktot + hi * 16;
  const __bf16* const hbase  = hSt + (size_t)mA * H_ + hi * 16;

  const int nktX = Kx >> 5;               // K tiles in the x region
  const int nktH = (Ktot - Kx) >> 5;      // K tiles in the h region (16)

  for (int t = 0; t < S_; ++t) {
    // per-batch-row sequence position (implements TF reverse_seq for bw)
    const int posA = (dir == 0) ? t : ((t < lenA) ? (lenA - 1 - t) : 0);

    v8f acc0, acc1, acc2, acc3;
#pragma unroll
    for (int r = 0; r < 8; ++r) { acc0[r] = 0.f; acc1[r] = 0.f; acc2[r] = 0.f; acc3[r] = 0.f; }

    // ---- x-region K tiles: pure pointer-increment loads + WMMA -------------
    const __bf16* ap  = Xin + ((size_t)posA * B_ + mA) * Kx + hi * 16;
    const __bf16* wp0 = wbase0;
    const __bf16* wp1 = wbase1;
    const __bf16* wp2 = wbase2;
    const __bf16* wp3 = wbase3;
    for (int kt = 0; kt < nktX; ++kt) {
      Frag16 af;
      af.q[0] = *(const uint4*)(ap);
      af.q[1] = *(const uint4*)(ap + 8);
      Frag16 b0, b1, b2, b3;
      b0.q[0] = *(const uint4*)(wp0);  b0.q[1] = *(const uint4*)(wp0 + 8);
      b1.q[0] = *(const uint4*)(wp1);  b1.q[1] = *(const uint4*)(wp1 + 8);
      b2.q[0] = *(const uint4*)(wp2);  b2.q[1] = *(const uint4*)(wp2 + 8);
      b3.q[0] = *(const uint4*)(wp3);  b3.q[1] = *(const uint4*)(wp3 + 8);
      acc0 = wmma_bf16(af.v, b0.v, acc0);
      acc1 = wmma_bf16(af.v, b1.v, acc1);
      acc2 = wmma_bf16(af.v, b2.v, acc2);
      acc3 = wmma_bf16(af.v, b3.v, acc3);
      ap += 32; wp0 += 32; wp1 += 32; wp2 += 32; wp3 += 32;
    }

    // ---- h-region K tiles (recurrent part) ---------------------------------
    const __bf16* hp = hbase;
    for (int kt = 0; kt < nktH; ++kt) {
      Frag16 af;
      af.q[0] = *(const uint4*)(hp);
      af.q[1] = *(const uint4*)(hp + 8);
      Frag16 b0, b1, b2, b3;
      b0.q[0] = *(const uint4*)(wp0);  b0.q[1] = *(const uint4*)(wp0 + 8);
      b1.q[0] = *(const uint4*)(wp1);  b1.q[1] = *(const uint4*)(wp1 + 8);
      b2.q[0] = *(const uint4*)(wp2);  b2.q[1] = *(const uint4*)(wp2 + 8);
      b3.q[0] = *(const uint4*)(wp3);  b3.q[1] = *(const uint4*)(wp3 + 8);
      acc0 = wmma_bf16(af.v, b0.v, acc0);
      acc1 = wmma_bf16(af.v, b1.v, acc1);
      acc2 = wmma_bf16(af.v, b2.v, acc2);
      acc3 = wmma_bf16(af.v, b3.v, acc3);
      hp += 32; wp0 += 32; wp1 += 32; wp2 += 32; wp3 += 32;
    }

    // Gate math. C layout: VGPR r -> M = mtile*16 + hi*8 + r, N = n.
#pragma unroll
    for (int r = 0; r < 8; ++r) {
      const int m = mtile * 16 + hi * 8 + r;
      float ii = sigmoidf_(acc0[r] + bi);
      float jj = tanhf_(acc1[r] + bj);
      float ff = sigmoidf_(acc2[r] + bff);
      float oo = sigmoidf_(acc3[r] + bo);
      float cn = ff * c_r[r] + ii * jj;
      float hn = oo * tanhf_(cn);
      const bool act = t < len_r[r];
      const int opos = act ? ((dir == 0) ? t : (len_r[r] - 1 - t)) : t;
      Out[((size_t)opos * B_ + m) * (2 * H_) + colOff + n] =
          act ? (__bf16)hn : (__bf16)0.0f;
      if (act) {
        c_r[r] = cn;
        hSt[(size_t)m * H_ + n] = (__bf16)hn;   // frozen past length
      }
    }

    __threadfence();
    dir_barrier(bar, 16);
  }
}

// ---------------------------------------------------------------------------
// 4) Dense(25) + softmax. One wave per (b,s) position; lanes 0..24 = classes.
// ---------------------------------------------------------------------------
__global__ __launch_bounds__(256) void logits_softmax_kernel(
    const __bf16* __restrict__ h1,      // [S][64][1024]
    const float* __restrict__ wd,       // [1024][25]
    const float* __restrict__ bd,       // [25]
    float* __restrict__ out) {          // [B][S][25]
  const int wave = threadIdx.x >> 5, lane = threadIdx.x & 31;
  const int pos = blockIdx.x * 8 + wave;   // = b*S + s
  const int b = pos >> 8, s = pos & 255;
  const __bf16* hp = h1 + ((size_t)s * B_ + b) * (2 * H_);
  const int c = (lane < C_) ? lane : (C_ - 1);
  float acc = bd[c];
  for (int k = 0; k < 2 * H_; ++k)
    acc = fmaf((float)hp[k], wd[k * C_ + c], acc);

  float lv = (lane < C_) ? acc : -INFINITY;
  float mx = lv;
#pragma unroll
  for (int m = 16; m; m >>= 1) mx = fmaxf(mx, __shfl_xor(mx, m, 32));
  float e = (lane < C_) ? __expf(acc - mx) : 0.0f;
  float ssum = e;
#pragma unroll
  for (int m = 16; m; m >>= 1) ssum += __shfl_xor(ssum, m, 32);
  if (lane < C_)
    out[((size_t)b * S_ + s) * C_ + lane] = e / ssum;
}

// ---------------------------------------------------------------------------
extern "C" void kernel_launch(void* const* d_in, const int* in_sizes, int n_in,
                              void* d_out, int out_size, void* d_ws, size_t ws_size,
                              hipStream_t stream) {
  (void)in_sizes; (void)n_in; (void)out_size; (void)ws_size;
  const int*   ids   = (const int*)d_in[0];
  const int*   lens  = (const int*)d_in[1];
  const float* emb   = (const float*)d_in[2];
  const float* w_fw0 = (const float*)d_in[3];
  const float* b_fw0 = (const float*)d_in[4];
  const float* w_bw0 = (const float*)d_in[5];
  const float* b_bw0 = (const float*)d_in[6];
  const float* w_fw1 = (const float*)d_in[7];
  const float* b_fw1 = (const float*)d_in[8];
  const float* w_bw1 = (const float*)d_in[9];
  const float* b_bw1 = (const float*)d_in[10];
  const float* wd    = (const float*)d_in[11];
  const float* bd    = (const float*)d_in[12];
  float* out = (float*)d_out;

  char* ws = (char*)d_ws;
  size_t off = 0;
  auto take = [&](size_t bytes) -> char* {
    char* p = ws + off;
    off = (off + bytes + 255) & ~(size_t)255;
    return p;
  };

  unsigned* bars = (unsigned*)take(1024);                 // 4 dirs x 256B slots
  __bf16* hstate = (__bf16*)take((size_t)4 * B_ * H_ * 2);
  const size_t zeroBytes = off;                           // zero bars + h-states
  __bf16* x0pad = (__bf16*)take((size_t)S_ * B_ * EPAD * 2);
  __bf16* h0cat = (__bf16*)take((size_t)S_ * B_ * 2 * H_ * 2);
  __bf16* h1cat = (__bf16*)take((size_t)S_ * B_ * 2 * H_ * 2);
  __bf16* wt0f  = (__bf16*)take((size_t)NG * K0TOT * 2);
  __bf16* wt0b  = (__bf16*)take((size_t)NG * K0TOT * 2);
  __bf16* wt1f  = (__bf16*)take((size_t)NG * K1TOT * 2);
  __bf16* wt1b  = (__bf16*)take((size_t)NG * K1TOT * 2);

  hipMemsetAsync(d_ws, 0, zeroBytes, stream);

  embed_pad_kernel<<<S_ * B_, 64, 0, stream>>>(ids, emb, x0pad);

  wconv_kernel<<<NG, 256, 0, stream>>>(w_fw0, wt0f, E_, EPAD, K0TOT);
  wconv_kernel<<<NG, 256, 0, stream>>>(w_bw0, wt0b, E_, EPAD, K0TOT);
  wconv_kernel<<<NG, 256, 0, stream>>>(w_fw1, wt1f, 2 * H_, 2 * H_, K1TOT);
  wconv_kernel<<<NG, 256, 0, stream>>>(w_bw1, wt1b, 2 * H_, 2 * H_, K1TOT);

  unsigned* barFw0 = bars + 0;
  unsigned* barBw0 = bars + 64;
  unsigned* barFw1 = bars + 128;
  unsigned* barBw1 = bars + 192;
  __bf16* hFw0 = hstate;
  __bf16* hBw0 = hstate + (size_t)1 * B_ * H_;
  __bf16* hFw1 = hstate + (size_t)2 * B_ * H_;
  __bf16* hBw1 = hstate + (size_t)3 * B_ * H_;

  lstm_layer_kernel<<<32, 256, 0, stream>>>(x0pad, wt0f, wt0b, b_fw0, b_bw0, lens,
                                            hFw0, hBw0, barFw0, barBw0, h0cat,
                                            EPAD, K0TOT);
  lstm_layer_kernel<<<32, 256, 0, stream>>>(h0cat, wt1f, wt1b, b_fw1, b_bw1, lens,
                                            hFw1, hBw1, barFw1, barBw1, h1cat,
                                            2 * H_, K1TOT);

  logits_softmax_kernel<<<(B_ * S_) / 8, 256, 0, stream>>>(h1cat, wd, bd, out);
}
